// ParametricEncoder_49108656063045
// MI455X (gfx1250) — matrix operand order repaired and checked by reading, not compiled, not run on hardware
//
#include <hip/hip_runtime.h>
#include <math.h>

// ---------------- problem constants ----------------
#define BATCH 16
#define NPTS  1024
#define CIN   6
#define EMBED 64

typedef __attribute__((ext_vector_type(16))) __bf16 v16bf;
typedef __attribute__((ext_vector_type(8)))  __bf16 v8bf;
typedef __attribute__((ext_vector_type(8)))  float  v8f;

__device__ __forceinline__ float gelu_f(float x) {
    return 0.5f * x * (1.0f + erff(x * 0.70710678118654752440f));
}

// ============================================================
// f32 -> bf16 copy (weights, just-in-time per res block)
// ============================================================
__global__ __launch_bounds__(256)
void f32_to_bf16(const float* __restrict__ src, __bf16* __restrict__ dst, int n)
{
    int i = blockIdx.x * 256 + threadIdx.x;
    if (i < n) dst[i] = (__bf16)src[i];
}

// ============================================================
// Embedding: x = feats^T * W (6 -> 64), GroupNorm(8) + GELU, fused in LDS.
// One block per (b, group). Output pf [B][64][1024].
// ============================================================
__global__ __launch_bounds__(256)
void embed_gn_gelu(const float* __restrict__ feats,   // [B][6][1024]
                   const float* __restrict__ W,       // [64][6]
                   const float* __restrict__ gamma,
                   const float* __restrict__ beta,
                   float* __restrict__ pf)            // [B][64][1024]
{
    int b  = blockIdx.x >> 3;
    int gr = blockIdx.x & 7;
    __shared__ float buf[8192];
    __shared__ float sh[256], sh2[256];
    float s = 0.f, s2 = 0.f;
    for (int i = threadIdx.x; i < 8192; i += 256) {
        int ch = i >> 10;
        int n  = i & 1023;
        int e  = gr * 8 + ch;
        float acc = 0.f;
#pragma unroll
        for (int c = 0; c < CIN; ++c)
            acc += feats[(b * CIN + c) * NPTS + n] * W[e * CIN + c];
        buf[i] = acc;
        s += acc; s2 += acc * acc;
    }
    sh[threadIdx.x] = s; sh2[threadIdx.x] = s2;
    __syncthreads();
    for (int st = 128; st > 0; st >>= 1) {
        if (threadIdx.x < st) { sh[threadIdx.x] += sh[threadIdx.x + st]; sh2[threadIdx.x] += sh2[threadIdx.x + st]; }
        __syncthreads();
    }
    float mu = sh[0] * (1.f / 8192.f);
    float var = sh2[0] * (1.f / 8192.f) - mu * mu;
    float is = rsqrtf(var + 1e-5f);
    for (int i = threadIdx.x; i < 8192; i += 256) {
        int ch = i >> 10;
        int e  = gr * 8 + ch;
        int n  = i & 1023;
        float v = (buf[i] - mu) * is * gamma[e] + beta[e];
        pf[((size_t)b * EMBED + e) * NPTS + n] = gelu_f(v);
    }
}

// ============================================================
// Farthest point sampling: one block per batch, sequential argmax loop.
// ============================================================
__global__ __launch_bounds__(256)
void fps_kernel(const float* __restrict__ xyz, int N, int npoint, int* __restrict__ fidx)
{
    int b = blockIdx.x;
    __shared__ float dist[1024];
    __shared__ float sval[256];
    __shared__ int   sidx[256];
    __shared__ int   sFar;
    const float* p = xyz + (size_t)b * N * 3;
    for (int i = threadIdx.x; i < N; i += 256) dist[i] = 1e10f;
    if (threadIdx.x == 0) sFar = 0;
    __syncthreads();
    for (int it = 0; it < npoint; ++it) {
        int far = sFar;
        if (threadIdx.x == 0) fidx[b * npoint + it] = far;
        float cx = p[far * 3 + 0], cy = p[far * 3 + 1], cz = p[far * 3 + 2];
        float best = -1.f; int bidx = 0;
        for (int i = threadIdx.x; i < N; i += 256) {
            float dx = p[i * 3 + 0] - cx, dy = p[i * 3 + 1] - cy, dz = p[i * 3 + 2] - cz;
            float d = dx * dx + dy * dy + dz * dz;
            float nd = fminf(dist[i], d);
            dist[i] = nd;
            if (nd > best) { best = nd; bidx = i; }
        }
        sval[threadIdx.x] = best; sidx[threadIdx.x] = bidx;
        __syncthreads();
        for (int st = 128; st > 0; st >>= 1) {
            if (threadIdx.x < st) {
                float v2 = sval[threadIdx.x + st]; int i2 = sidx[threadIdx.x + st];
                if (v2 > sval[threadIdx.x] || (v2 == sval[threadIdx.x] && i2 < sidx[threadIdx.x])) {
                    sval[threadIdx.x] = v2; sidx[threadIdx.x] = i2;
                }
            }
            __syncthreads();
        }
        if (threadIdx.x == 0) sFar = sidx[0];
        __syncthreads();
    }
}

// ============================================================
// kNN: one block per (b,g); LDS distances, K sequential argmins.
// ============================================================
__global__ __launch_bounds__(256)
void knn_kernel(const float* __restrict__ xyz, int N, const int* __restrict__ fidx,
                int G, int K, int* __restrict__ kidx)
{
    int bg = blockIdx.x;
    int b = bg / G, g = bg % G;
    __shared__ float dist[1024];
    __shared__ float sval[256];
    __shared__ int   sidx[256];
    const float* p = xyz + (size_t)b * N * 3;
    int q = fidx[b * G + g];
    float qx = p[q * 3 + 0], qy = p[q * 3 + 1], qz = p[q * 3 + 2];
    for (int i = threadIdx.x; i < N; i += 256) {
        float dx = p[i * 3 + 0] - qx, dy = p[i * 3 + 1] - qy, dz = p[i * 3 + 2] - qz;
        dist[i] = dx * dx + dy * dy + dz * dz;
    }
    __syncthreads();
    for (int j = 0; j < K; ++j) {
        float best = 3.4e38f; int bidx = 0;
        for (int i = threadIdx.x; i < N; i += 256) {
            float d = dist[i];
            if (d < best) { best = d; bidx = i; }
        }
        sval[threadIdx.x] = best; sidx[threadIdx.x] = bidx;
        __syncthreads();
        for (int st = 128; st > 0; st >>= 1) {
            if (threadIdx.x < st) {
                float v2 = sval[threadIdx.x + st]; int i2 = sidx[threadIdx.x + st];
                if (v2 < sval[threadIdx.x] || (v2 == sval[threadIdx.x] && i2 < sidx[threadIdx.x])) {
                    sval[threadIdx.x] = v2; sidx[threadIdx.x] = i2;
                }
            }
            __syncthreads();
        }
        if (threadIdx.x == 0) { kidx[(size_t)(b * G + g) * K + j] = sidx[0]; dist[sidx[0]] = 3.4e38f; }
        __syncthreads();
    }
}

// ============================================================
// Normalized relative coords + gather of next-stage coords.
// ============================================================
__global__ __launch_bounds__(256)
void relnorm_kernel(const float* __restrict__ xyz, int N, const int* __restrict__ fidx,
                    const int* __restrict__ kidx, int G, int K,
                    float* __restrict__ rel, float* __restrict__ coordsNext, int total)
{
    int i = blockIdx.x * 256 + threadIdx.x;
    if (i >= total) return;
    int b = i / G, g = i % G;
    const float* p = xyz + (size_t)b * N * 3;
    int q = fidx[b * G + g];
    float lx = p[q * 3 + 0], ly = p[q * 3 + 1], lz = p[q * 3 + 2];
    coordsNext[(size_t)(b * G + g) * 3 + 0] = lx;
    coordsNext[(size_t)(b * G + g) * 3 + 1] = ly;
    coordsNext[(size_t)(b * G + g) * 3 + 2] = lz;
    float mx = 0.f, my = 0.f, mz = 0.f;
    const int* ki = kidx + (size_t)(b * G + g) * K;
    for (int k = 0; k < K; ++k) {
        int id = ki[k];
        mx = fmaxf(mx, fabsf(p[id * 3 + 0] - lx));
        my = fmaxf(my, fabsf(p[id * 3 + 1] - ly));
        mz = fmaxf(mz, fabsf(p[id * 3 + 2] - lz));
    }
    mx = fmaxf(mx, 1e-5f); my = fmaxf(my, 1e-5f); mz = fmaxf(mz, 1e-5f);
    float* ro = rel + ((size_t)(b * G + g) * K) * 3;
    for (int k = 0; k < K; ++k) {
        int id = ki[k];
        ro[k * 3 + 0] = (p[id * 3 + 0] - lx) / mx;
        ro[k * 3 + 1] = (p[id * 3 + 1] - ly) / my;
        ro[k * 3 + 2] = (p[id * 3 + 2] - lz) / mz;
    }
}

// ============================================================
// xq (channel-last [b][m=g*K+k][ch]) = gathered features + sinusoidal PE.
// Writes f32 master (residual path) + bf16 shadow (GEMM input).
// ============================================================
__global__ __launch_bounds__(256)
void build_xq(const float* __restrict__ pf, int Cin, int Nprev,
              const int* __restrict__ fidx, const int* __restrict__ kidx,
              const float* __restrict__ rel, int cd, int G, int K,
              float* __restrict__ X, __bf16* __restrict__ Xb, size_t total)
{
    size_t i = (size_t)blockIdx.x * 256 + threadIdx.x;
    if (i >= total) return;
    int ch = (int)(i % cd);
    size_t t = i / cd;
    int k_ = (int)(t % K); t /= K;
    int g  = (int)(t % G);
    int b  = (int)(t / G);

    float base;
    if (ch < Cin)
        base = pf[((size_t)b * Cin + ch) * Nprev + kidx[((size_t)(b * G + g)) * K + k_]];
    else
        base = pf[((size_t)b * Cin + (ch - Cin)) * Nprev + fidx[b * G + g]];

    int F = (cd + 5) / 6;
    int axis = ch / (2 * F);
    int rem  = ch - axis * 2 * F;
    int f = (rem < F) ? rem : rem - F;
    float de = expf(((float)f / (float)F) * 4.605170185988091f);  // 100^(f/F)
    float dv = 16.0f * rel[(((size_t)(b * G + g)) * K + k_) * 3 + axis] / de;
    float pe = (rem < F) ? sinf(dv) : cosf(dv);
    float v = base + pe;
    X[i]  = v;              // i == ((b*G+g)*K + k_)*cd + ch  == [b][m][ch]
    Xb[i] = (__bf16)v;
}

// ============================================================
// WMMA GEMM (channel-last, pure bf16 operands):
//   Out[b][m][h] = sum_c X[b][m][c] * W[h][c]
// Loop body: 6x global_load_b128 (8 bf16 each) + 2x v_wmma_f32_16x16x32_bf16.
// Each wave: one 16-col m tile x two 16-row h tiles (B tile loaded once).
// Lane layouts per CDNA5 ISA 7.12.2 (wave32); shuffles are reg-placement only.
// ============================================================
__global__ __launch_bounds__(128)
void gemm_wmma_bf16(const __bf16* __restrict__ W,     // [Hc][Cc] bf16
                    const __bf16* __restrict__ Xin,   // [B][M][Cc] bf16
                    float* __restrict__ Out,          // [B][M][Hc] f32
                    int Hc, int Cc, int M)
{
    int lane = threadIdx.x;               // 0..31
    int m0   = (blockIdx.x * 4 + threadIdx.y) * 16;
    int h0   = blockIdx.y * 32;           // two h tiles: h0, h0+16
    int b    = blockIdx.z;
    int half = lane >> 4;                 // 0 or 1
    int r    = lane & 15;

    const __bf16* xrow  = Xin + (size_t)b * M * Cc + (size_t)(m0 + r) * Cc + half * 8;
    const __bf16* w0row = W + (size_t)(h0 + r) * Cc + half * 8;
    const __bf16* w1row = W + (size_t)(h0 + 16 + r) * Cc + half * 8;

    v8f acc0 = {}, acc1 = {};
    for (int c0 = 0; c0 < Cc; c0 += 32) {
        // B tile (32x16 KxN): chunks K = half*8+0..7 and 16+half*8+0..7
        v8bf bl = *(const v8bf*)(xrow + c0);
        v8bf bh = *(const v8bf*)(xrow + c0 + 16);
        v16bf bm = __builtin_shufflevector(bl, bh, 0, 1, 2, 3, 4, 5, 6, 7,
                                                   8, 9, 10, 11, 12, 13, 14, 15);
        // A tiles (16x32 MxK): chunks K = half*8+0..7 and 16+half*8+0..7
        v8bf a0l = *(const v8bf*)(w0row + c0);
        v8bf a0h = *(const v8bf*)(w0row + c0 + 16);
        v16bf a0 = __builtin_shufflevector(a0l, a0h, 0, 1, 2, 3, 4, 5, 6, 7,
                                                     8, 9, 10, 11, 12, 13, 14, 15);
        v8bf a1l = *(const v8bf*)(w1row + c0);
        v8bf a1h = *(const v8bf*)(w1row + c0 + 16);
        v16bf a1 = __builtin_shufflevector(a1l, a1h, 0, 1, 2, 3, 4, 5, 6, 7,
                                                     8, 9, 10, 11, 12, 13, 14, 15);

        acc0 = __builtin_amdgcn_wmma_f32_16x16x32_bf16(false, a0, false, bm, (short)0, acc0, false, false);
        acc1 = __builtin_amdgcn_wmma_f32_16x16x32_bf16(false, a1, false, bm, (short)0, acc1, false, false);
    }

    // D: VGPR j -> (h = h0 + half*8 + j, m = m0 + r): contiguous b128 pairs
    float* orow = Out + (size_t)b * M * Hc + (size_t)(m0 + r) * Hc;
    float4 o;
    o = make_float4(acc0[0], acc0[1], acc0[2], acc0[3]);
    ((float4*)(orow + h0 + half * 8))[0] = o;
    o = make_float4(acc0[4], acc0[5], acc0[6], acc0[7]);
    ((float4*)(orow + h0 + half * 8))[1] = o;
    o = make_float4(acc1[0], acc1[1], acc1[2], acc1[3]);
    ((float4*)(orow + h0 + 16 + half * 8))[0] = o;
    o = make_float4(acc1[4], acc1[5], acc1[6], acc1[7]);
    ((float4*)(orow + h0 + 16 + half * 8))[1] = o;
}

// ============================================================
// GroupNorm stats (channel-last): one block per (b,gr).
// ============================================================
__global__ __launch_bounds__(256)
void gn_stats(const float* __restrict__ X, float* __restrict__ stats, int C, int M)
{
    int bg = blockIdx.x;
    int b = bg >> 3, gr = bg & 7;
    int Cg = C >> 3;
    int sh_cg = __builtin_ctz(Cg);
    const float* base = X + (size_t)b * M * C + gr * Cg;
    size_t L = (size_t)M * Cg;
    float s = 0.f, s2 = 0.f;
    for (size_t i = threadIdx.x; i < L; i += 256) {
        size_t m = i >> sh_cg;
        int cl = (int)(i & (Cg - 1));
        float v = base[m * C + cl];
        s += v; s2 += v * v;
    }
    __shared__ float sh[256], sh2[256];
    sh[threadIdx.x] = s; sh2[threadIdx.x] = s2;
    __syncthreads();
    for (int st = 128; st > 0; st >>= 1) {
        if (threadIdx.x < st) { sh[threadIdx.x] += sh[threadIdx.x + st]; sh2[threadIdx.x] += sh2[threadIdx.x + st]; }
        __syncthreads();
    }
    if (threadIdx.x == 0) {
        float inv = 1.0f / (float)L;
        float mu = sh[0] * inv;
        float var = sh2[0] * inv - mu * mu;
        stats[bg * 2 + 0] = mu;
        stats[bg * 2 + 1] = rsqrtf(var + 1e-5f);
    }
}

// Hb = bf16( gelu(gn(H)) )  -- only bf16 shadow consumed downstream
__global__ __launch_bounds__(256)
void gn_gelu(const float* __restrict__ H, __bf16* __restrict__ Hb,
             const float* __restrict__ stats,
             const float* __restrict__ gamma, const float* __restrict__ beta,
             int C, int M, size_t total)
{
    size_t i = (size_t)blockIdx.x * 256 + threadIdx.x;
    if (i >= total) return;
    int c = (int)(i % C);
    int b = (int)(i / ((size_t)M * C));
    int gr = c / (C >> 3);
    float mu = stats[(b * 8 + gr) * 2 + 0];
    float is = stats[(b * 8 + gr) * 2 + 1];
    float v = (H[i] - mu) * is * gamma[c] + beta[c];
    Hb[i] = (__bf16)gelu_f(v);
}

// X = gelu(gn(Y) + X); also refresh bf16 shadow Xb for the next block's GEMM
__global__ __launch_bounds__(256)
void gn_add_gelu(const float* __restrict__ Y, float* __restrict__ X,
                 __bf16* __restrict__ Xb,
                 const float* __restrict__ stats,
                 const float* __restrict__ gamma, const float* __restrict__ beta,
                 int C, int M, size_t total)
{
    size_t i = (size_t)blockIdx.x * 256 + threadIdx.x;
    if (i >= total) return;
    int c = (int)(i % C);
    int b = (int)(i / ((size_t)M * C));
    int gr = c / (C >> 3);
    float mu = stats[(b * 8 + gr) * 2 + 0];
    float is = stats[(b * 8 + gr) * 2 + 1];
    float v = (Y[i] - mu) * is * gamma[c] + beta[c];
    float o = gelu_f(v + X[i]);
    X[i]  = o;
    Xb[i] = (__bf16)o;
}

// max over K neighbors: X [b][g*K+k][c] -> pf [b][c][g]
__global__ __launch_bounds__(256)
void maxpool_k(const float* __restrict__ X, float* __restrict__ pf,
               int cd, int G, int K, int total)
{
    int i = blockIdx.x * 256 + threadIdx.x;
    if (i >= total) return;
    int c = i % cd;
    int g = (i / cd) % G;
    int b = i / (cd * G);
    const float* p = X + ((size_t)(b * G + g) * K) * cd + c;
    float m = p[0];
    for (int k = 1; k < K; ++k) m = fmaxf(m, p[(size_t)k * cd]);
    pf[((size_t)b * cd + c) * G + g] = m;
}

// ============================================================
// Host orchestration
// ============================================================
extern "C" void kernel_launch(void* const* d_in, const int* in_sizes, int n_in,
                              void* d_out, int out_size, void* d_ws, size_t ws_size,
                              hipStream_t stream)
{
    (void)in_sizes; (void)n_in; (void)out_size; (void)ws_size;
    const float* coords0 = (const float*)d_in[0];    // [16,1024,3]
    const float* feats   = (const float*)d_in[1];    // [16,6,1024]
    const float* embW    = (const float*)d_in[2];
    const float* embG    = (const float*)d_in[3];
    const float* embB    = (const float*)d_in[4];
    int pi = 5;

    // ---- workspace layout (floats); ~85M floats (~340 MB) ----
    float* ws = (float*)d_ws;
    size_t off = 0;
    float*  Xbuf = ws + off; off += 25165824;          // [B][M][cd] f32 master
    float*  Ybuf = ws + off; off += 25165824;          // GEMM2 out f32
    float*  Hbuf = ws + off; off += 12582912;          // GEMM1 out f32
    __bf16* Xbf  = (__bf16*)(ws + off); off += 12582912;  // bf16 shadow of X
    __bf16* Hbf  = (__bf16*)(ws + off); off += 6291456;   // bf16 shadow of H
    __bf16* Wbf  = (__bf16*)(ws + off); off += 262144;    // per-block weight arena
    float*  pfA  = ws + off; off += 1048576;           // [B][C][N]
    float*  pfB  = ws + off; off += 1048576;
    float*  cA   = ws + off; off += 49152;
    float*  cB   = ws + off; off += 49152;
    float*  relb = ws + off; off += 589824;            // [B][G][K][3] max
    float*  stat = ws + off; off += 256;               // [B*8][2]
    int*    fidx = (int*)(ws + off); off += 8192;
    int*    kidx = (int*)(ws + off); off += 196608;

    embed_gn_gelu<<<BATCH * 8, 256, 0, stream>>>(feats, embW, embG, embB, pfA);

    const float* coordsCur = coords0;
    int   Ncur = NPTS;
    float* pfCur = pfA;
    float* pfNext = pfB;
    int   Cin = EMBED;
    float* coordBufs[2] = { cA, cB };

    const int stG[3]  = { 512, 256, 128 };
    const int stK[3]  = { 24, 24, 16 };
    const int stCD[3] = { 128, 256, 512 };
    const int stNB[3] = { 2, 2, 1 };
    dim3 gemmBlk(32, 4, 1);

    for (int s = 0; s < 3; ++s) {
        int G = stG[s], K = stK[s], cd = stCD[s], nb = stNB[s];
        int hid = cd / 2;
        int M = G * K;
        float* cNxt = coordBufs[s & 1];

        fps_kernel<<<BATCH, 256, 0, stream>>>(coordsCur, Ncur, G, fidx);
        knn_kernel<<<BATCH * G, 256, 0, stream>>>(coordsCur, Ncur, fidx, G, K, kidx);
        relnorm_kernel<<<(BATCH * G + 255) / 256, 256, 0, stream>>>(
            coordsCur, Ncur, fidx, kidx, G, K, relb, cNxt, BATCH * G);

        size_t total = (size_t)BATCH * cd * M;
        build_xq<<<(unsigned)((total + 255) / 256), 256, 0, stream>>>(
            pfCur, Cin, Ncur, fidx, kidx, relb, cd, G, K, Xbuf, Xbf, total);

        for (int blk = 0; blk < nb; ++blk) {
            const float* W1 = (const float*)d_in[pi++];
            const float* g1 = (const float*)d_in[pi++];
            const float* b1 = (const float*)d_in[pi++];
            const float* W2 = (const float*)d_in[pi++];
            const float* g2 = (const float*)d_in[pi++];
            const float* b2 = (const float*)d_in[pi++];

            // JIT weight conversion into the bf16 arena
            int n1 = hid * cd;              // W1 [hid][cd]
            int n2 = cd * hid;              // W2 [cd][hid]
            __bf16* W1b = Wbf;
            __bf16* W2b = Wbf + n1;
            f32_to_bf16<<<(n1 + 255) / 256, 256, 0, stream>>>(W1, W1b, n1);
            f32_to_bf16<<<(n2 + 255) / 256, 256, 0, stream>>>(W2, W2b, n2);

            dim3 grid1(M / 64, hid / 32, BATCH);
            gemm_wmma_bf16<<<grid1, gemmBlk, 0, stream>>>(W1b, Xbf, Hbuf, hid, cd, M);
            gn_stats<<<BATCH * 8, 256, 0, stream>>>(Hbuf, stat, hid, M);
            size_t th = (size_t)BATCH * hid * M;
            gn_gelu<<<(unsigned)((th + 255) / 256), 256, 0, stream>>>(
                Hbuf, Hbf, stat, g1, b1, hid, M, th);

            dim3 grid2(M / 64, cd / 32, BATCH);
            gemm_wmma_bf16<<<grid2, gemmBlk, 0, stream>>>(W2b, Hbf, Ybuf, cd, hid, M);
            gn_stats<<<BATCH * 8, 256, 0, stream>>>(Ybuf, stat, cd, M);
            gn_add_gelu<<<(unsigned)((total + 255) / 256), 256, 0, stream>>>(
                Ybuf, Xbuf, Xbf, stat, g2, b2, cd, M, total);
        }

        int tp = BATCH * cd * G;
        maxpool_k<<<(tp + 255) / 256, 256, 0, stream>>>(Xbuf, pfNext, cd, G, K, tp);

        coordsCur = cNxt;
        Ncur = G;
        Cin = cd;
        float* t = pfCur; pfCur = pfNext; pfNext = t;
    }

    // ---- outputs: coords [16,128,3] then features [16,512,128] ----
    float* out = (float*)d_out;
    hipMemcpyAsync(out, coordsCur, (size_t)BATCH * 128 * 3 * sizeof(float),
                   hipMemcpyDeviceToDevice, stream);
    hipMemcpyAsync(out + BATCH * 128 * 3, pfCur, (size_t)BATCH * 512 * 128 * sizeof(float),
                   hipMemcpyDeviceToDevice, stream);
}
